// BitNet_68083821576572
// MI455X (gfx1250) — matrix operand order repaired and checked by reading, not compiled, not run on hardware
//
#include <hip/hip_runtime.h>
#include <hip/hip_bf16.h>
#include <math.h>

typedef __attribute__((ext_vector_type(16))) _Float16     v16h;
typedef __attribute__((ext_vector_type(8)))  _Float16     v8h;
typedef __attribute__((ext_vector_type(8)))  float        v8f;
typedef __attribute__((ext_vector_type(4)))  unsigned int v4u;
typedef __attribute__((ext_vector_type(8)))  int          v8i;
typedef __attribute__((ext_vector_type(4)))  int          v4i;

#define NPTS  32
#define BATCH 8

#if __has_builtin(__builtin_amdgcn_tensor_load_to_lds)
#define HAVE_TDM 1
#else
#define HAVE_TDM 0
#endif
#if __has_include(<hip/amd_detail/amd_gfx1250_TDM.h>)
#define TDM_6ARG 1
#else
#define TDM_6ARG 0
#endif

// ---------------------------------------------------------------------------
// Weight prep: f32 OIHW -> f16 [K][Cout], tap-major K (k = tap*Cin + cin).
// Gives the conv kernels a TDM-friendly 2D matrix with row stride Cout.
// ---------------------------------------------------------------------------
__global__ void prep_weights(const float* __restrict__ w, _Float16* __restrict__ out,
                             int cinLog2, int Cout)
{
    const int Cin = 1 << cinLog2;
    const int K = Cin * 9;
    int idx = blockIdx.x * blockDim.x + threadIdx.x;
    if (idx >= K * Cout) return;
    int oc = idx % Cout, k = idx / Cout;
    int t = k >> cinLog2, ci = k & (Cin - 1);
    out[idx] = (_Float16)w[((size_t)oc * Cin + ci) * 9 + t];
}

// ---------------------------------------------------------------------------
// 3x3 SAME conv, implicit GEMM on the WMMA pipe. Activations f16 NHWC.
//   M = H*W pixels, N = Cout, K = Cin*9 tap-major (Cin = power of two).
// A-fragment 8-slot runs are 8 contiguous channels at one pixel
// (one global_load_b128 + one uniform bounds check per run).
// Weight panels (KC rows x 16*NT couts) are DMA'd into LDS by the Tensor
// Data Mover (wave 0 issues tensor_load_to_lds, waits TENSORcnt, then the
// workgroup barrier releases the other waves); TDM zero-fills the K tail
// via tensor_dim1 = K - kc.
// ---------------------------------------------------------------------------
template<int NT>
__global__ __launch_bounds__(256)
void conv3x3_wmma_nhwc(const _Float16* __restrict__ in,
                       const _Float16* __restrict__ wt,   // f16 [K][Cout]
                       const float* __restrict__ bias,
                       void* __restrict__ outv,
                       int cinLog2, int H, int W, int Cout,
                       int relu, int outF32)
{
    constexpr int KC = 256;          // K-rows staged per LDS panel
    constexpr int RW = 16 * NT;      // LDS row width (couts per block)
    __shared__ alignas(32) _Float16 sB[KC * RW];
    const int Cin = 1 << cinLog2;
    const int K   = Cin * 9;
    const int tid = threadIdx.x;
    const int wave = tid >> 5;
    const int lane = tid & 31;
    const int coutBase = blockIdx.y * RW;
    const int b = blockIdx.z;
    const size_t HW = (size_t)H * W;

    const _Float16* inb = in + (size_t)b * HW * Cin;
    const int tileM = blockIdx.x * 128 + wave * 16;
    const int m  = tileM + (lane & 15);
    const int h0 = m / W;
    const int w0 = m % W;
    const int ksel = (lane >> 4) << 3;   // A-layout K split across lane halves

    __builtin_prefetch(inb + ((size_t)h0 * W + w0) * Cin, 0, 1);

    v8f acc[NT];
#pragma unroll
    for (int nt = 0; nt < NT; ++nt) acc[nt] = v8f{};

    for (int kc = 0; kc < K; kc += KC) {
#if HAVE_TDM
        if (wave == 0) {
            // Tensor DMA descriptor: 2D tile [KC rows x RW] of f16 matrix
            // [K][Cout], tile origin (kc, coutBase), row stride Cout.
            unsigned ldsAddr = (unsigned)(size_t)&sB[0];
            unsigned long long ga =
                (unsigned long long)(size_t)(wt + (size_t)kc * Cout + coutBase);
            v4u g0 = { 1u,                                   // count=1 (valid)
                       ldsAddr,                              // lds_addr
                       (unsigned)ga,                         // global_addr lo
                       ((unsigned)(ga >> 32) & 0x01FFFFFFu)  // global_addr hi
                           | 0x80000000u };                  // type=2 (image)
            int validRows = K - kc;                          // TDM zero-fills rest
            v8i g1 = { 0x00010000,                           // data_size=1 (2B)
                       (int)((RW & 0xFFFF) << 16),           // tensor_dim0
                       (int)((validRows & 0xFFFF) << 16),    // tensor_dim1 lo
                       (int)(RW << 16),                      // tile_dim0
                       (int)KC,                              // tile_dim1
                       Cout,                                 // dim0 stride (elems)
                       0, 0 };
            v4i gz = { 0, 0, 0, 0 };
#if TDM_6ARG
            v8i g4 = { 0, 0, 0, 0, 0, 0, 0, 0 };
            __builtin_amdgcn_tensor_load_to_lds(g0, g1, gz, gz, g4, 0);
#else
            __builtin_amdgcn_tensor_load_to_lds(g0, g1, gz, gz, 0);
#endif
            __builtin_amdgcn_s_wait_tensorcnt(0);
        }
        __syncthreads();
#else
        for (int i = tid; i < KC * RW; i += 256) {
            int r = i / RW, c = i % RW;
            int k = kc + r;
            sB[i] = (k < K) ? wt[(size_t)k * Cout + coutBase + c] : (_Float16)0.f;
        }
        __syncthreads();
#endif

        int kend = K - kc; if (kend > KC) kend = KC;
        for (int k0 = 0; k0 < kend; k0 += 32) {
            // A fragment: slots 0-7 <-> k = kkA+0..7, slots 8-15 <-> kkA+16..23.
            // Runs are 8-aligned; taps are Cin(>=16)-aligned -> never straddle.
            v8h ra = {}, rb = {};
            const int kkA = kc + k0 + ksel;
            if (kkA < K) {
                int t = kkA >> cinLog2, ci = kkA & (Cin - 1);
                int hh = h0 + t / 3 - 1, ww = w0 + t % 3 - 1;
                if (hh >= 0 && hh < H && ww >= 0 && ww < W)
                    ra = *(const v8h*)(inb + ((size_t)hh * W + ww) * Cin + ci);
            }
            const int kkB = kkA + 16;
            if (kkB < K) {
                int t = kkB >> cinLog2, ci = kkB & (Cin - 1);
                int hh = h0 + t / 3 - 1, ww = w0 + t % 3 - 1;
                if (hh >= 0 && hh < H && ww >= 0 && ww < W)
                    rb = *(const v8h*)(inb + ((size_t)hh * W + ww) * Cin + ci);
            }
            v16h a = __builtin_shufflevector(ra, rb, 0, 1, 2, 3, 4, 5, 6, 7,
                                                     8, 9, 10, 11, 12, 13, 14, 15);
            const int kb = (k0 + ksel) * RW + (lane & 15);     // B column n=lane%16, table-order k base
#pragma unroll
            for (int nt = 0; nt < NT; ++nt) {
                v16h bm;
#pragma unroll
                for (int i = 0; i < 16; ++i)
                    bm[i] = sB[kb + ((i < 8) ? i : i + 8) * RW + nt * 16];
                acc[nt] = __builtin_amdgcn_wmma_f32_16x16x32_f16(
                              false, a, false, bm, (short)0, acc[nt], false, false);
            }
        }
        __syncthreads();
    }

    // D layout: lanes 0-15: N=lane, VGPR r -> M=r ; lanes 16-31: N=lane-16, M=r+8.
    const int n = lane & 15;
    float bi[NT];
#pragma unroll
    for (int nt = 0; nt < NT; ++nt) bi[nt] = bias[coutBase + nt * 16 + n];
    const int mBase = tileM + ((lane >> 4) << 3);
#pragma unroll
    for (int r = 0; r < 8; ++r) {
        size_t o = ((size_t)b * HW + (mBase + r)) * Cout + coutBase + n;
#pragma unroll
        for (int nt = 0; nt < NT; ++nt) {
            float v = acc[nt][r] + bi[nt];
            if (relu) v = fmaxf(v, 0.f);
            if (outF32) ((float*)outv)[o + nt * 16] = v;
            else        ((_Float16*)outv)[o + nt * 16] = (_Float16)v;
        }
    }
}

// ---------------------------------------------------------------------------
// First conv: Cin=1 (f32 NCHW input), Cout=16, K=9 (one WMMA), out f16 NHWC.
__global__ __launch_bounds__(256)
void conv1_wmma(const float* __restrict__ x, const float* __restrict__ w,
                const float* __restrict__ bias, _Float16* __restrict__ out,
                int H, int W)
{
    __shared__ alignas(32) _Float16 sB[32 * 16];
    for (int i = threadIdx.x; i < 32 * 16; i += 256) {
        int k = i >> 4, n = i & 15;
        sB[i] = (_Float16)((k < 9) ? w[n * 9 + k] : 0.f);
    }
    __syncthreads();

    const int tid = threadIdx.x, wave = tid >> 5, lane = tid & 31;
    const int b = blockIdx.z;
    const size_t HW = (size_t)H * W;
    const float* xb = x + (size_t)b * HW;
    const int tileM = blockIdx.x * 128 + wave * 16;
    const int m  = tileM + (lane & 15);
    const int h0 = m / W, w0 = m % W;
    const int ksel = (lane >> 4) << 3;

    v16h a;
#pragma unroll
    for (int i = 0; i < 16; ++i) {
        int k = ksel + ((i < 8) ? i : i + 8);
        float v = 0.f;
        if (k < 9) {
            int hh = h0 + k / 3 - 1, ww = w0 + k % 3 - 1;
            if (hh >= 0 && hh < H && ww >= 0 && ww < W)
                v = xb[(size_t)hh * W + ww];
        }
        a[i] = (_Float16)v;
    }
    v16h bm;
#pragma unroll
    for (int i = 0; i < 16; ++i)
        bm[i] = sB[(ksel + ((i < 8) ? i : i + 8)) * 16 + (lane & 15)];
    v8f acc = {};
    acc = __builtin_amdgcn_wmma_f32_16x16x32_f16(false, a, false, bm,
                                                 (short)0, acc, false, false);

    const int n = lane & 15;
    const float bi = bias[n];
    const int mBase = tileM + ((lane >> 4) << 3);
#pragma unroll
    for (int r = 0; r < 8; ++r) {
        float v = fmaxf(acc[r] + bi, 0.f);
        out[((size_t)b * HW + (mBase + r)) * 16 + n] = (_Float16)v;
    }
}

// ---------------------------------------------------------------------------
__global__ void maxpool2_nhwc(const _Float16* __restrict__ in,
                              _Float16* __restrict__ out,
                              int B_, int H, int W, int C)
{
    int Ho = H >> 1, Wo = W >> 1;
    size_t total = (size_t)B_ * Ho * Wo * C;
    size_t idx = (size_t)blockIdx.x * blockDim.x + threadIdx.x;
    if (idx >= total) return;
    int c = (int)(idx % C);
    size_t q = idx / C;
    int wo = (int)(q % Wo); q /= Wo;
    int ho = (int)(q % Ho);
    int b  = (int)(q / Ho);
    const _Float16* p = in + (((size_t)b * H + 2 * ho) * W + 2 * wo) * C + c;
    size_t rs = (size_t)W * C;
    float v = fmaxf(fmaxf((float)p[0], (float)p[C]),
                    fmaxf((float)p[rs], (float)p[rs + C]));
    out[idx] = (_Float16)v;
}

// ---------------------------------------------------------------------------
// Soft-argmax over 64x64, floored. Logits are f32 NHWC (C=32). Block per (b,p).
__global__ __launch_bounds__(256)
void softargmax64(const float* __restrict__ logits, float* __restrict__ pts)
{
    __shared__ float red[256];
    const int bp = blockIdx.x;
    const float* lg = logits + (size_t)(bp >> 5) * 4096 * NPTS + (bp & 31);
    const int tid = threadIdx.x;

    float lm = -3.4e38f;
    for (int i = tid; i < 4096; i += 256) lm = fmaxf(lm, lg[(size_t)i * NPTS]);
    red[tid] = lm; __syncthreads();
    for (int s = 128; s > 0; s >>= 1) { if (tid < s) red[tid] = fmaxf(red[tid], red[tid + s]); __syncthreads(); }
    float mx = red[0]; __syncthreads();

    float s0 = 0.f, s1 = 0.f, s2 = 0.f;
    for (int i = tid; i < 4096; i += 256) {
        float e = expf(lg[(size_t)i * NPTS] - mx);
        s0 += e; s1 += e * (float)(i >> 6); s2 += e * (float)(i & 63);
    }
    red[tid] = s0; __syncthreads();
    for (int s = 128; s > 0; s >>= 1) { if (tid < s) red[tid] += red[tid + s]; __syncthreads(); }
    float S = red[0]; __syncthreads();
    red[tid] = s1; __syncthreads();
    for (int s = 128; s > 0; s >>= 1) { if (tid < s) red[tid] += red[tid + s]; __syncthreads(); }
    float SH = red[0]; __syncthreads();
    red[tid] = s2; __syncthreads();
    for (int s = 128; s > 0; s >>= 1) { if (tid < s) red[tid] += red[tid + s]; __syncthreads(); }
    if (tid == 0) {
        float SW = red[0];
        pts[bp * 2 + 0] = floorf(SH / S);
        pts[bp * 2 + 1] = floorf(SW / S);
    }
}

// ---------------------------------------------------------------------------
// patches[b, p*C+c, i, j] = fm_nhwc[b, clip(2h+i), clip(2w+j), c]  (f16 -> f32)
__global__ void extract_patches(const _Float16* __restrict__ fm,
                                const float* __restrict__ pts,
                                float* __restrict__ patch,
                                int C, int H, int W)
{
    int total = BATCH * NPTS * C * 4;
    int idx = blockIdx.x * blockDim.x + threadIdx.x;
    if (idx >= total) return;
    int j = idx & 1, i = (idx >> 1) & 1;
    int c = (idx >> 2) % C;
    int p = ((idx >> 2) / C) % NPTS;
    int b = idx / (4 * C * NPTS);
    int sh = (int)(pts[(b * NPTS + p) * 2 + 0] * 2.f);
    int sw = (int)(pts[(b * NPTS + p) * 2 + 1] * 2.f);
    int hi = min(max(sh + i, 0), H - 1);
    int wi = min(max(sw + j, 0), W - 1);
    patch[idx] = (float)fm[(((size_t)b * H + hi) * W + wi) * C + c];
}

// ---------------------------------------------------------------------------
// Grouped (groups=32) 3x3 SAME conv on 2x2 patches (f32, tiny workload).
__global__ void gconv2x2(const float* __restrict__ in, const float* __restrict__ w,
                         const float* __restrict__ bias, float* __restrict__ out,
                         int CinG, int CoutG, int relu)
{
    int total = BATCH * NPTS * CoutG * 4;
    int idx = blockIdx.x * blockDim.x + threadIdx.x;
    if (idx >= total) return;
    int j  = idx & 1, i = (idx >> 1) & 1;
    int co = (idx >> 2) % CoutG;
    int g  = ((idx >> 2) / CoutG) % NPTS;
    int b  = idx / (4 * CoutG * NPTS);
    const float* ig = in + ((size_t)(b * NPTS + g) * CinG) * 4;
    const float* wg = w + (size_t)(g * CoutG + co) * CinG * 9;
    float acc = bias[g * CoutG + co];
    for (int cin = 0; cin < CinG; ++cin) {
        const float* ic = ig + cin * 4;
        const float* wc = wg + cin * 9;
#pragma unroll
        for (int kh = 0; kh < 3; ++kh) {
            int ih = i + kh - 1;
            if (ih < 0 || ih > 1) continue;
#pragma unroll
            for (int kw = 0; kw < 3; ++kw) {
                int iw = j + kw - 1;
                if (iw < 0 || iw > 1) continue;
                acc += ic[ih * 2 + iw] * wc[kh * 3 + kw];
            }
        }
    }
    if (relu) acc = fmaxf(acc, 0.f);
    out[idx] = acc;
}

// ---------------------------------------------------------------------------
__global__ void softargmax2x2_update(const float* __restrict__ g3,
                                     const float* __restrict__ ptsIn,
                                     float* __restrict__ ptsOut)
{
    int t = blockIdx.x * blockDim.x + threadIdx.x;
    if (t >= BATCH * NPTS) return;
    const float* l = g3 + (size_t)t * 4;
    float mx = fmaxf(fmaxf(l[0], l[1]), fmaxf(l[2], l[3]));
    float e0 = expf(l[0] - mx), e1 = expf(l[1] - mx);
    float e2 = expf(l[2] - mx), e3 = expf(l[3] - mx);
    float S  = e0 + e1 + e2 + e3;
    float eh = (e2 + e3) / S;
    float ew = (e1 + e3) / S;
    float sh = (float)(int)(ptsIn[t * 2 + 0] * 2.f);
    float sw = (float)(int)(ptsIn[t * 2 + 1] * 2.f);
    ptsOut[t * 2 + 0] = sh + floorf(eh);
    ptsOut[t * 2 + 1] = sw + floorf(ew);
}

// ---------------------------------------------------------------------------
extern "C" void kernel_launch(void* const* d_in, const int* in_sizes, int n_in,
                              void* d_out, int out_size, void* d_ws, size_t ws_size,
                              hipStream_t stream)
{
    (void)in_sizes; (void)n_in; (void)out_size; (void)ws_size;
    const float* x = (const float*)d_in[0];
    char* ws = (char*)d_ws;

    // Workspace layout (bytes); activations are f16 NHWC.
    _Float16* T0  = (_Float16*)(ws + (size_t)0);            // 67,108,864
    _Float16* FM1 = (_Float16*)(ws + (size_t)67108864);     // 67,108,864
    _Float16* T1  = (_Float16*)(ws + (size_t)134217728);    // 33,554,432
    _Float16* FM2 = (_Float16*)(ws + (size_t)167772160);    // 33,554,432
    _Float16* FM3 = (_Float16*)(ws + (size_t)201326592);    // 16,777,216
    float*    LOG = (float*)   (ws + (size_t)218103808);    //  4,194,304 (f32 logits)
    char*     sm  = ws + (size_t)222298112;
    float* PTS   = (float*)(sm);
    float* PATCH = (float*)(sm + 4096);
    float* G1    = (float*)(sm + 4096 + 524288);
    float* G2    = (float*)(sm + 4096 + 2 * 524288);
    float* G3    = (float*)(sm + 4096 + 3 * 524288);
    _Float16* WTS = (_Float16*)(ws + (size_t)226492416);    // ~660 KB f16 weights

    // --- weight prep: f32 OIHW -> f16 [K][Cout] tap-major ---
    struct WL { int wi, cl2, Cout; size_t off; };
    WL wl[8] = {
        {  3, 4,  16,      0 }, {  5, 4,  32,   2304 },
        {  7, 5,  32,   6912 }, {  9, 5,  64,  16128 },
        { 11, 6,  64,  34560 }, { 13, 6, 128,  71424 },
        { 15, 7, 128, 145152 }, { 17, 7,  32, 292608 },
    };
    for (int l = 0; l < 8; ++l) {
        int total = (9 << wl[l].cl2) * wl[l].Cout;
        prep_weights<<<(total + 255) / 256, 256, 0, stream>>>(
            (const float*)d_in[wl[l].wi], WTS + wl[l].off, wl[l].cl2, wl[l].Cout);
    }

    auto conv = [&](const _Float16* in, int layer, void* out,
                    int H, int W, int relu, int outF32) {
        int cl2 = wl[layer].cl2, Cout = wl[layer].Cout;
        const _Float16* wt = WTS + wl[layer].off;
        const float* bias = (const float*)d_in[wl[layer].wi + 1];
        int NTv = (Cout >= 32) ? 2 : 1;
        dim3 grid((H * W) / 128, Cout / (16 * NTv), BATCH);
        if (NTv == 2)
            conv3x3_wmma_nhwc<2><<<grid, 256, 0, stream>>>(
                in, wt, bias, out, cl2, H, W, Cout, relu, outF32);
        else
            conv3x3_wmma_nhwc<1><<<grid, 256, 0, stream>>>(
                in, wt, bias, out, cl2, H, W, Cout, relu, outF32);
    };
    auto pool = [&](const _Float16* in, _Float16* out, int H, int W, int C) {
        size_t total = (size_t)BATCH * (H / 2) * (W / 2) * C;
        maxpool2_nhwc<<<(unsigned)((total + 255) / 256), 256, 0, stream>>>(
            in, out, BATCH, H, W, C);
    };

    // --- encoder ---
    {   // Cin=1 special conv: x f32 NCHW -> T0 f16 NHWC (C=16), relu
        dim3 grid((512 * 512) / 128, 1, BATCH);
        conv1_wmma<<<grid, 256, 0, stream>>>(x, (const float*)d_in[1],
                                             (const float*)d_in[2], T0, 512, 512);
    }
    conv(T0, 0, FM1, 512, 512, 1, 0);
    pool(FM1, T0, 512, 512, 16);
    conv(T0, 1, T1,  256, 256, 1, 0);
    conv(T1, 2, FM2, 256, 256, 1, 0);
    pool(FM2, T0, 256, 256, 32);
    conv(T0, 3, T1,  128, 128, 1, 0);
    conv(T1, 4, FM3, 128, 128, 1, 0);
    pool(FM3, T0, 128, 128, 64);

    // --- lowest head ---
    conv(T0, 5, T1,  64, 64, 1, 0);
    conv(T1, 6, T0,  64, 64, 1, 0);
    conv(T0, 7, LOG, 64, 64, 0, 1);
    softargmax64<<<BATCH * NPTS, 256, 0, stream>>>(LOG, PTS);

    // --- coarse-to-fine refinement (reversed seq2 with reversed fms) ---
    struct Stage { const _Float16* fm; int C, H, W, wi, mid; };
    Stage stages[3] = {
        { FM3, 64, 128, 128, 31, 128 },   // seq2_params[2]
        { FM2, 32, 256, 256, 25,  64 },   // seq2_params[1]
        { FM1, 16, 512, 512, 19,  32 },   // seq2_params[0]
    };
    for (int s = 0; s < 3; ++s) {
        const Stage& st = stages[s];
        int tE = BATCH * NPTS * st.C * 4;
        extract_patches<<<(tE + 255) / 256, 256, 0, stream>>>(st.fm, PTS, PATCH,
                                                              st.C, st.H, st.W);
        int tM = BATCH * NPTS * st.mid * 4;
        gconv2x2<<<(tM + 255) / 256, 256, 0, stream>>>(PATCH,
            (const float*)d_in[st.wi],     (const float*)d_in[st.wi + 1], G1, st.C,  st.mid, 1);
        gconv2x2<<<(tM + 255) / 256, 256, 0, stream>>>(G1,
            (const float*)d_in[st.wi + 2], (const float*)d_in[st.wi + 3], G2, st.mid, st.mid, 1);
        int tL = BATCH * NPTS * 4;
        gconv2x2<<<(tL + 255) / 256, 256, 0, stream>>>(G2,
            (const float*)d_in[st.wi + 4], (const float*)d_in[st.wi + 5], G3, st.mid, 1, 0);
        float* outPts = (s == 2) ? (float*)d_out : PTS;
        softargmax2x2_update<<<1, 256, 0, stream>>>(G3, PTS, outPts);
    }
}
